// dispatcher_42580305773211
// MI455X (gfx1250) — matrix operand order repaired and checked by reading, not compiled, not run on hardware
//
#include <hip/hip_runtime.h>
#include <hip/hip_bf16.h>

// ---------------- problem constants ----------------
#define NROWS   131072      // B*T
#define DDIM    256
#define HDIM    512
#define NLAYER  4
#define EPSV    1e-8f
#define ROWS_PER_TILE 15    // 16-row tiles overlapped by 1 row (pair coverage)
#define NTILES  ((NROWS + ROWS_PER_TILE - 1) / ROWS_PER_TILE)   // 8739
#define SSTR    264         // bf16/row (528B): 16B aligned, 4-dword bank phase
#define H1STR   520         // bf16/row (1040B): 16B aligned, 4-dword bank phase
#define CSTR    260         // f32/row (1040B) -- cb rows coincide with stg2 rows
#define GRP     8           // n-tiles per group (8 independent WMMA chains)
#define WPB     4           // waves per block

// per-wave LDS slice layout (bytes)
#define STG_B   (16 * SSTR * 2)     //  8448  nxt/inv bf16 (K=256)
#define STG2_B  (16 * H1STR * 2)    // 16640  h1 bf16 (K=512); cb aliases exactly
#define NX1_B   (256 * 4)           //  1024  fp32 overlap row (m == 15)
#define GB_B    (16 * 4)            //    64  per-row S/norm scale
#define WAVE_LDS (STG_B + STG2_B + NX1_B + GB_B)   // 26176
#define SMEM_BYTES (WPB * WAVE_LDS)                // 104704 -> 3 blocks/WGP

typedef __attribute__((ext_vector_type(16))) __bf16 v16bf;
typedef __attribute__((ext_vector_type(8)))  float  v8f;

union ABu { uint4 u[2]; v16bf v; };
union B4u { __bf16 h[4]; uint2 u2; };

static __device__ __forceinline__ v8f wmma_bf(v16bf a, v16bf b, v8f c) {
  return __builtin_amdgcn_wmma_f32_16x16x32_bf16(false, a, false, b,
                                                 (short)0, c, false, false);
}
static __device__ __forceinline__ v8f wmma_bf0(v16bf a, v16bf b) {
  v8f z = {};                              // C = inline 0
  return __builtin_amdgcn_wmma_f32_16x16x32_bf16(false, a, false, b,
                                                 (short)0, z, false, false);
}

// A fragment (16x32 bf16) from row-major LDS stage (row stride `stride` bf16).
static __device__ __forceinline__ v16bf ld_afrag(const __bf16* base, int stride,
                                                 int kb, int l16, int half) {
  ABu t;
  const __bf16* p = base + l16 * stride + kb + half * 8;
  t.u[0] = *(const uint4*)(p);
  t.u[1] = *(const uint4*)(p + 16);
  return t.v;
}

// B fragment (32x16 bf16): lane holds column n (=its weight row), 16
// bf16 at K = kbase + (lane/16)*8 + {0..7, 16..23}.
static __device__ __forceinline__ v16bf ld_bfrag(const __bf16* kbase, int half) {
  ABu t;
  const __bf16* p = kbase + half * 8;
  t.u[0] = *(const uint4*)(p);
  t.u[1] = *(const uint4*)(p + 16);
  return t.v;
}

// Group GEMM: 8 accumulator chains sharing one A fragment per K-step.
template <int K32, int WSTRIDE>
static __device__ __forceinline__ void gemm_group(const __bf16* astage, int astride,
                                                  const __bf16* wg,
                                                  int l16, int half, v8f acc[GRP]) {
  v16bf a_cur = ld_afrag(astage, astride, 0, l16, half);
  v16bf b_cur[GRP];
#pragma unroll
  for (int u = 0; u < GRP; ++u)
    b_cur[u] = ld_bfrag(wg + (long)u * 16 * WSTRIDE, half);

#pragma unroll
  for (int ks = 0; ks < K32; ++ks) {
    v16bf a_nxt = a_cur;
    v16bf b_nxt[GRP];
    if (ks + 1 < K32) {
      a_nxt = ld_afrag(astage, astride, (ks + 1) * 32, l16, half);
#pragma unroll
      for (int u = 0; u < GRP; ++u)
        b_nxt[u] = ld_bfrag(wg + (long)u * 16 * WSTRIDE + (ks + 1) * 32, half);
    }
#pragma unroll
    for (int u = 0; u < GRP; ++u)
      acc[u] = (ks == 0) ? wmma_bf0(a_cur, b_cur[u])
                         : wmma_bf(a_cur, b_cur[u], acc[u]);
    a_cur = a_nxt;
#pragma unroll
    for (int u = 0; u < GRP; ++u) b_cur[u] = b_nxt[u];
  }
}

// ---------------- prep: fp32 weights -> bf16 copies, zero loss accum ----------------
extern "C" __global__ void prep_weights(const float* __restrict__ W1,
                                        const float* __restrict__ W2,
                                        const float* __restrict__ Wc,
                                        __bf16* __restrict__ W1b,
                                        __bf16* __restrict__ W2b,
                                        __bf16* __restrict__ Wcb,
                                        float* __restrict__ dsum) {
  int i = blockIdx.x * 256 + threadIdx.x;
  if (i < HDIM * DDIM) { W1b[i] = (__bf16)W1[i]; W2b[i] = (__bf16)W2[i]; }
  if (i < NLAYER * DDIM * DDIM) Wcb[i] = (__bf16)Wc[i];
  if (i < DDIM) dsum[i] = 0.f;
}

// ---------------- main fused chain: one wave = one 16-row tile ----------------
extern "C" __global__ __launch_bounds__(128)
void fused_chain(const float* __restrict__ x,
                 const float* __restrict__ b1,
                 const float* __restrict__ b2,
                 const float* __restrict__ bc,
                 const __bf16* __restrict__ W1b,   // (512,256)
                 const __bf16* __restrict__ W2b,   // (256,512)
                 const __bf16* __restrict__ Wcb,   // (4,256,256)
                 float* __restrict__ out_total,
                 float* __restrict__ out_nxt,     // doubles as fp32 chain state
                 float* __restrict__ dsum) {
  extern __shared__ char smem[];
  const int lane = threadIdx.x & 31;
  const int wid  = threadIdx.x >> 5;
  const int half = lane >> 4;
  const int l16  = lane & 15;
  const int m_lane = 8 * half;
  const int tile = blockIdx.x * WPB + wid;
  if (tile >= NTILES) return;
  const long s = (long)tile * ROWS_PER_TILE;

  char* wbase = smem + wid * WAVE_LDS;
  __bf16* stg  = (__bf16*)wbase;                    // nxt / inv bf16 (K=256)
  __bf16* stg2 = (__bf16*)(wbase + STG_B);          // h1 bf16 (K=512)
  float*  cb   = (float*)(wbase + STG_B);           // loss 'out' f32 (aliases stg2)
  float*  nx1  = (float*)(wbase + STG_B + STG2_B);  // fp32 chain row m==15
  float*  gb   = (float*)(wbase + STG_B + STG2_B + NX1_B);

  float dacc[8];
#pragma unroll
  for (int q = 0; q < 8; ++q) dacc[q] = 0.f;

  for (int k = 0; k < NLAYER; ++k) {
    // ---- stage current nxt as bf16 A operand (vectorized) ----
    // chain state: k==0 -> x ; else rows 0..14 -> out_nxt, row 15 -> nx1 (LDS)
#pragma unroll
    for (int m = 0; m < 16; ++m) {
      long r = s + m;
      if (r > NROWS - 1) r = NROWS - 1;   // clamped rows: dead, never written
      __bf16* sr = stg + m * SSTR;
#pragma unroll
      for (int q = 0; q < 2; ++q) {
        int e = lane + 32 * q;
        float4 t;
        if (k == 0)       t = ((const float4*)(x + r * DDIM))[e];
        else if (m == 15) t = ((const float4*)nx1)[e];
        else              t = ((const float4*)(out_nxt + r * DDIM))[e];
        B4u b;
        b.h[0] = (__bf16)t.x; b.h[1] = (__bf16)t.y;
        b.h[2] = (__bf16)t.z; b.h[3] = (__bf16)t.w;
        *(uint2*)(sr + e * 4) = b.u2;
      }
    }

    // ---- GEMM1: h1 = nxt@W1^T + b1 -> stg2 (16x512 bf16) ----
    for (int g = 0; g < 4; ++g) {
      int nb = g * GRP;
      v8f acc[GRP];
      gemm_group<8, DDIM>(stg, SSTR, W1b + (long)(nb * 16 + l16) * DDIM,
                          l16, half, acc);
#pragma unroll
      for (int u = 0; u < GRP; ++u) {
        float bv = b1[(nb + u) * 16 + l16];
#pragma unroll
        for (int v = 0; v < 8; ++v)
          stg2[(v + m_lane) * H1STR + (nb + u) * 16 + l16] = (__bf16)(acc[u][v] + bv);
      }
    }

    // ---- GEMM2: inv = h1@W2^T + b2 (K=512); epilogue: chain update + stage inv ----
    for (int g = 0; g < 2; ++g) {
      int nb = g * GRP;
      v8f acc[GRP];
      gemm_group<16, HDIM>(stg2, H1STR, W2b + (long)(nb * 16 + l16) * HDIM,
                           l16, half, acc);
#pragma unroll
      for (int u = 0; u < GRP; ++u) {
        float bv = b2[(nb + u) * 16 + l16];
#pragma unroll
        for (int v = 0; v < 8; ++v) {
          int m = v + m_lane, n = (nb + u) * 16 + l16;
          float iv = acc[u][v] + bv;
          stg[m * SSTR + n] = (__bf16)iv;
          long r = s + m;
          if (m == 15) {
            long rc = (r > NROWS - 1) ? NROWS - 1 : r;
            float gprev = (k == 0) ? x[rc * DDIM + n] : nx1[n];
            nx1[n] = gprev - iv;
          } else if (r < NROWS) {
            float gprev = (k == 0) ? x[r * DDIM + n] : out_nxt[r * DDIM + n];
            out_nxt[r * DDIM + n] = gprev - iv;
          }
        }
      }
    }

    // ---- loss GEMM: out = inv @ Wc_k^T + bc_k; cb (aliases stg2) + S/normsq ----
    const __bf16* Wk = Wcb + (long)k * DDIM * DDIM;
    const float* bck = bc + k * DDIM;
    float ps[8], pq[8];
#pragma unroll
    for (int v = 0; v < 8; ++v) { ps[v] = 0.f; pq[v] = 0.f; }
    for (int g = 0; g < 2; ++g) {
      int nb = g * GRP;
      v8f acc[GRP];
      gemm_group<8, DDIM>(stg, SSTR, Wk + (long)(nb * 16 + l16) * DDIM,
                          l16, half, acc);
#pragma unroll
      for (int u = 0; u < GRP; ++u) {
        float bv = bck[(nb + u) * 16 + l16];
#pragma unroll
        for (int v = 0; v < 8; ++v) {
          float o = acc[u][v] + bv;
          cb[(v + m_lane) * CSTR + (nb + u) * 16 + l16] = o;
          ps[v] += o;
          pq[v] += o * o;
        }
      }
    }
    // reduce row sums across the 16 lanes of each half-wave
#pragma unroll
    for (int v = 0; v < 8; ++v)
#pragma unroll
      for (int d = 1; d < 16; d <<= 1) {
        ps[v] += __shfl_xor(ps[v], d, 32);
        pq[v] += __shfl_xor(pq[v], d, 32);
      }
    if (l16 == 0) {
#pragma unroll
      for (int v = 0; v < 8; ++v)
        gb[v + m_lane] = ps[v] * __builtin_amdgcn_rcpf(fmaxf(sqrtf(pq[v]), EPSV));
    }

    // per-column |c[m+1]-c[m]| accumulation; tile covers pairs m=0..14
    long mlim = (long)(NROWS - 2) - s;
    if (mlim > 14) mlim = 14;
#pragma unroll
    for (int q = 0; q < 8; ++q) {
      int j = lane + 32 * q;
      float o0 = cb[j];
      float pc = o0 * gb[0] * __builtin_amdgcn_rcpf(fmaxf(fabsf(o0) * 16.0f, EPSV));
#pragma unroll
      for (int m = 1; m < 16; ++m) {
        float om = cb[m * CSTR + j];
        float cm = om * gb[m] * __builtin_amdgcn_rcpf(fmaxf(fabsf(om) * 16.0f, EPSV));
        if ((long)(m - 1) <= mlim) dacc[q] += fabsf(cm - pc);
        pc = cm;
      }
    }
  }

  // ---- outputs: out_nxt already holds final nxt; total = x - nxt (b128) ----
#pragma unroll
  for (int m = 0; m < 15; ++m) {
    long r = s + m;
    if (r < NROWS) {
      const float4* xr = (const float4*)(x + r * DDIM);
      const float4* nr = (const float4*)(out_nxt + r * DDIM);
      float4* ot = (float4*)(out_total + r * DDIM);
#pragma unroll
      for (int q = 0; q < 2; ++q) {
        int e = lane + 32 * q;
        float4 nv = nr[e];
        float4 xv = xr[e];
        float4 tv;
        tv.x = xv.x - nv.x; tv.y = xv.y - nv.y;
        tv.z = xv.z - nv.z; tv.w = xv.w - nv.w;
        ot[e] = tv;
      }
    }
  }

#pragma unroll
  for (int q = 0; q < 8; ++q)
    atomicAdd(&dsum[lane + 32 * q], dacc[q]);
}

// ---------------- broadcast loss vector to (D,D) ----------------
extern "C" __global__ void bcast_loss(const float* __restrict__ dsum,
                                      float* __restrict__ loss) {
  int i = blockIdx.x * 256 + threadIdx.x;
  if (i < DDIM * DDIM) loss[i] = dsum[i & (DDIM - 1)];
}

extern "C" void kernel_launch(void* const* d_in, const int* in_sizes, int n_in,
                              void* d_out, int out_size, void* d_ws, size_t ws_size,
                              hipStream_t stream) {
  (void)in_sizes; (void)n_in; (void)out_size; (void)ws_size;
  const float* x  = (const float*)d_in[0];
  const float* W1 = (const float*)d_in[1];
  const float* b1 = (const float*)d_in[2];
  const float* W2 = (const float*)d_in[3];
  const float* b2 = (const float*)d_in[4];
  const float* Wc = (const float*)d_in[5];
  const float* bc = (const float*)d_in[6];

  char* ws = (char*)d_ws;
  __bf16* W1b = (__bf16*)(ws);                 // 512*256*2 = 262144 B
  __bf16* W2b = (__bf16*)(ws + 262144);        // 256*512*2
  __bf16* Wcb = (__bf16*)(ws + 524288);        // 4*256*256*2
  float*  dsum = (float*)(ws + 1048576);       // 256 floats

  float* out_total = (float*)d_out;
  float* out_nxt   = out_total + (long)NROWS * DDIM;
  float* out_loss  = out_nxt + (long)NROWS * DDIM;

  prep_weights<<<1024, 256, 0, stream>>>(W1, W2, Wc, W1b, W2b, Wcb, dsum);

  dim3 grid((NTILES + WPB - 1) / WPB);
  fused_chain<<<grid, 32 * WPB, SMEM_BYTES, stream>>>(x, b1, b2, bc, W1b, W2b, Wcb,
                                                      out_total, out_nxt, dsum);

  bcast_loss<<<256, 256, 0, stream>>>(dsum, out_loss);
}
